// CFConv_5909874999436
// MI455X (gfx1250) — compile-verified, matrix operand order checked by reading, not run on hardware
//
#include <hip/hip_runtime.h>

typedef __attribute__((ext_vector_type(16))) _Float16     v16h;
typedef __attribute__((ext_vector_type(2)))  __fp16       v2fp16;
typedef __attribute__((ext_vector_type(8)))  float        v8f;
typedef __attribute__((ext_vector_type(4)))  unsigned int v4u;
typedef __attribute__((ext_vector_type(8)))  int          v8i;
typedef __attribute__((ext_vector_type(4)))  int          v4i;

#define DIM     64
#define TILE_E  128         // edges per block (8 waves x 16 edges)
#define F32S    66          // f32 LDS row stride (TDM pads +2 dwords per 64-dword row)
#define HS      72          // f16 LDS row stride for the GEMM2 staging tile
#define WS_BYTES 16384      // 2 matrices * 8 frags * 2 chunks * 32 lanes * 16B

union Frag { v16h v; unsigned int u[8]; v4u q[2]; };

__device__ __forceinline__ unsigned int pack2h(float a, float b) {
    union { v2fp16 h; unsigned int u; } z;
    z.h = __builtin_amdgcn_cvt_pkrtz(a, b);   // v_cvt_pk_rtz_f16_f32
    return z.u;
}

// K index inside a 16-bit WMMA A/B fragment: VGPR-pair p, lane-half hf, K-step ks
__device__ __forceinline__ int fragK(int ks, int p, int hf) {
    return 32*ks + 16*(p >> 2) + 8*hf + 2*(p & 3);
}

// shifted softplus: 2*log(1+exp(0.5x)), linear passthrough when 0.5x > 14.
// Hardware transcendentals: v_exp_f32 / v_log_f32 (vs ~35-op precise OCML path).
__device__ __forceinline__ float ssp(float x) {
    float h = 0.5f * x;
    float s = 2.0f * __logf(1.0f + __expf(h));
    return (h > 14.0f) ? x : s;
}

// ---------------------------------------------------------------------------
// Setup: precompute per-lane WMMA B fragments for W1/W2 (identical for every
// wave in the grid). Layout: ws[m][frag][chunk][lane] of 16B -> b128 loads.
// ---------------------------------------------------------------------------
__global__ void build_bfrags(const float* __restrict__ W1,
                             const float* __restrict__ W2,
                             unsigned int* __restrict__ ws)
{
    const int lane = threadIdx.x & 31;
    const int l16  = lane & 15;
    const int hf   = lane >> 4;
    for (int m = 0; m < 2; ++m) {
        const float* W = m ? W2 : W1;
        #pragma unroll
        for (int nt = 0; nt < 4; ++nt) {
            #pragma unroll
            for (int ks = 0; ks < 2; ++ks) {
                const int f = nt*2 + ks;
                const int N = nt*16 + l16;
                unsigned int u[8];
                #pragma unroll
                for (int p = 0; p < 8; ++p) {
                    const int K = fragK(ks, p, hf);
                    u[p] = pack2h(W[K*DIM + N], W[(K+1)*DIM + N]);
                }
                v4u* base = (v4u*)(ws + (((m*8 + f)*2 + 0)*32 + lane)*4);
                base[0]  = (v4u){u[0], u[1], u[2], u[3]};
                v4u* hi  = (v4u*)(ws + (((m*8 + f)*2 + 1)*32 + lane)*4);
                hi[0]    = (v4u){u[4], u[5], u[6], u[7]};
            }
        }
    }
}

// ---------------------------------------------------------------------------
// Main kernel
// ---------------------------------------------------------------------------
template <bool USEWS>
__global__ __launch_bounds__(256)
void cfconv_wmma_kernel(const float* __restrict__ nodew,
                        const float* __restrict__ rbf,
                        const float* __restrict__ W1,
                        const float* __restrict__ b1,
                        const float* __restrict__ W2,
                        const float* __restrict__ b2,
                        const int*   __restrict__ src,
                        const int*   __restrict__ dst,
                        const unsigned int* __restrict__ ws,
                        float*       __restrict__ out,
                        int E)
{
    __shared__ float    ldsF32[TILE_E * F32S];   // 33.8 KB : TDM-staged rbf tile
    __shared__ _Float16 ldsH  [TILE_E * HS];     // 18.4 KB : softplus(GEMM1) tile

    const int t    = threadIdx.x;
    const int lane = t & 31;
    const int wave = t >> 5;        // 0..7 (contiguous 32 threads = one wave32)
    const int l16  = lane & 15;
    const int hf   = lane >> 4;

    const long long tileBase = (long long)blockIdx.x * TILE_E;

    // ---- Issue TDM: DMA this block's 128x64 f32 rbf tile into padded LDS ----
    if (wave == 0) {
        unsigned long long gaddr =
            (unsigned long long)(const void*)rbf + (unsigned long long)tileBase * (DIM * 4);
        unsigned int ldsoff = (unsigned int)(uintptr_t)&ldsF32[0];
        long long rem = (long long)E - tileBase;           // rows available (OOB -> 0)
        if (rem > 0x7fffffffLL) rem = 0x7fffffffLL;
        if (rem < 0)            rem = 0;
        const unsigned int td1 = (unsigned int)rem;

        v4u g0;
        g0[0] = 1u;                                        // count=1 user descriptor
        g0[1] = ldsoff;                                    // lds_addr
        g0[2] = (unsigned int)(gaddr & 0xffffffffu);       // global_addr[31:0]
        g0[3] = (unsigned int)((gaddr >> 32) & 0x01ffffffu)// global_addr[56:32]
              | (2u << 30);                                // type=2 (image)

        v8i g1;
        g1[0] = (int)((2u << 16)      // data_size = 4B
                    | (1u << 20)      // pad_enable
                    | (5u << 22)      // pad_interval: every 64 dwords (one row)
                    | (1u << 25));    // pad_amount: 2 dwords -> row stride 66
        g1[1] = (int)(64u << 16);                             // tensor_dim0[15:0]=64
        g1[2] = (int)(((td1 & 0xffffu) << 16));               // dim0[31:16]=0 | dim1 lo
        g1[3] = (int)((td1 >> 16) | (64u << 16));             // dim1 hi | tile_dim0=64
        g1[4] = (int)128u;                                    // tile_dim1=128, tile_dim2=0
        g1[5] = (int)64u;                                     // tensor_dim0_stride=64
        g1[6] = 0;
        g1[7] = 0;

        v4i g2 = (v4i){0, 0, 0, 0};
        v4i g3 = (v4i){0, 0, 0, 0};
        v8i g4 = (v8i){0, 0, 0, 0, 0, 0, 0, 0};   // extra group, zero-filled (6-arg form)
        __builtin_amdgcn_tensor_load_to_lds(g0, g1, g2, g3, g4, 0);
    }

    // ---- B fragments + biases (overlapped with the TDM transfer) ----
    Frag B1[4][2], B2[4][2];
    float bias1[4], bias2[4];
    if (USEWS) {
        const v4u* wsq = (const v4u*)ws;
        #pragma unroll
        for (int nt = 0; nt < 4; ++nt) {
            #pragma unroll
            for (int ks = 0; ks < 2; ++ks) {
                const int f = nt*2 + ks;
                B1[nt][ks].q[0] = wsq[((0*8 + f)*2 + 0)*32 + lane];
                B1[nt][ks].q[1] = wsq[((0*8 + f)*2 + 1)*32 + lane];
                B2[nt][ks].q[0] = wsq[((1*8 + f)*2 + 0)*32 + lane];
                B2[nt][ks].q[1] = wsq[((1*8 + f)*2 + 1)*32 + lane];
            }
        }
    } else {
        #pragma unroll
        for (int nt = 0; nt < 4; ++nt) {
            const int N = nt*16 + l16;
            #pragma unroll
            for (int ks = 0; ks < 2; ++ks) {
                #pragma unroll
                for (int p = 0; p < 8; ++p) {
                    const int K = fragK(ks, p, hf);
                    B1[nt][ks].u[p] = pack2h(W1[K*DIM + N], W1[(K+1)*DIM + N]);
                    B2[nt][ks].u[p] = pack2h(W2[K*DIM + N], W2[(K+1)*DIM + N]);
                }
            }
        }
    }
    #pragma unroll
    for (int nt = 0; nt < 4; ++nt) {
        bias1[nt] = b1[nt*16 + l16];
        bias2[nt] = b2[nt*16 + l16];
    }

    if (wave == 0) __builtin_amdgcn_s_wait_tensorcnt(0);
    __syncthreads();

    const int myrow = wave*16 + l16;

    // ---- A1 fragments: f32 pairs from padded LDS, packed-convert to f16 ----
    Frag A1[2];
    #pragma unroll
    for (int ks = 0; ks < 2; ++ks) {
        #pragma unroll
        for (int p = 0; p < 8; ++p) {
            const int K = fragK(ks, p, hf);
            const float2 f = *(const float2*)(ldsF32 + myrow*F32S + K);  // ds_load_b64
            A1[ks].u[p] = pack2h(f.x, f.y);
        }
    }

    // ---- GEMM1: acc = rbf_f16 @ W1_f16 + b1 (f32 accumulate) ----
    v8f acc[4];
    #pragma unroll
    for (int nt = 0; nt < 4; ++nt) {
        v8f c;
        #pragma unroll
        for (int i = 0; i < 8; ++i) c[i] = bias1[nt];
        c = __builtin_amdgcn_wmma_f32_16x16x32_f16(false, A1[0].v, false, B1[nt][0].v,
                                                   (short)0, c, false, false);
        c = __builtin_amdgcn_wmma_f32_16x16x32_f16(false, A1[1].v, false, B1[nt][1].v,
                                                   (short)0, c, false, false);
        acc[nt] = c;
    }

    // ---- softplus, restage as f16 (D layout -> row major) ----
    #pragma unroll
    for (int nt = 0; nt < 4; ++nt) {
        const int N = nt*16 + l16;
        #pragma unroll
        for (int r = 0; r < 8; ++r) {
            const int M = 8*hf + r;
            ldsH[(wave*16 + M)*HS + N] = (_Float16)ssp(acc[nt][r]);
        }
    }
    __syncthreads();

    // ---- A2 fragments + GEMM2: e = sp @ W2_f16 + b2 ----
    Frag A2[2];
    #pragma unroll
    for (int ks = 0; ks < 2; ++ks) {
        #pragma unroll
        for (int p = 0; p < 8; ++p) {
            const int K = fragK(ks, p, hf);
            A2[ks].u[p] = *(const unsigned int*)(ldsH + myrow*HS + K);
        }
    }

    v8f ef[4];
    #pragma unroll
    for (int nt = 0; nt < 4; ++nt) {
        v8f c;
        #pragma unroll
        for (int i = 0; i < 8; ++i) c[i] = bias2[nt];
        c = __builtin_amdgcn_wmma_f32_16x16x32_f16(false, A2[0].v, false, B2[nt][0].v,
                                                   (short)0, c, false, false);
        c = __builtin_amdgcn_wmma_f32_16x16x32_f16(false, A2[1].v, false, B2[nt][1].v,
                                                   (short)0, c, false, false);
        ef[nt] = c;
    }

    // ---- gather node_weight[src] (L2-resident), multiply, scatter-add ----
    #pragma unroll
    for (int r = 0; r < 8; ++r) {
        const int M = 8*hf + r;
        const long long edge = tileBase + wave*16 + M;
        if (edge < E) {
            const int s = src[edge];
            const int d = dst[edge];
            #pragma unroll
            for (int nt = 0; nt < 4; ++nt) {
                const int N = nt*16 + l16;
                float m  = nodew[(long long)s*DIM + N] * ef[nt][r];
                float* p = out + (long long)d*DIM + N;
                // native non-returning f32 atomic add, device scope (no CAS loop)
                asm volatile("global_atomic_add_f32 %0, %1, off scope:SCOPE_DEV"
                             :: "v"(p), "v"(m) : "memory");
            }
        }
    }
}

extern "C" void kernel_launch(void* const* d_in, const int* in_sizes, int n_in,
                              void* d_out, int out_size, void* d_ws, size_t ws_size,
                              hipStream_t stream) {
    const float* nodew = (const float*)d_in[0];
    const float* rbf   = (const float*)d_in[1];
    const float* W1    = (const float*)d_in[2];
    const float* b1    = (const float*)d_in[3];
    const float* W2    = (const float*)d_in[4];
    const float* b2    = (const float*)d_in[5];
    const int*   src   = (const int*)d_in[6];
    const int*   dst   = (const int*)d_in[7];
    float*       out   = (float*)d_out;
    unsigned int* ws   = (unsigned int*)d_ws;

    const int E = in_sizes[6];

    (void)hipMemsetAsync(d_out, 0, (size_t)out_size * sizeof(float), stream);

    const int tiles = (E + TILE_E - 1) / TILE_E;   // 12500 for E = 1.6M
    if (ws_size >= WS_BYTES) {
        build_bfrags<<<1, 32, 0, stream>>>(W1, W2, ws);
        cfconv_wmma_kernel<true><<<tiles, 256, 0, stream>>>(
            nodew, rbf, W1, b1, W2, b2, src, dst, ws, out, E);
    } else {
        cfconv_wmma_kernel<false><<<tiles, 256, 0, stream>>>(
            nodew, rbf, W1, b1, W2, b2, src, dst, ws, out, E);
    }
}
